// fftformer_cross_24953759990233
// MI455X (gfx1250) — matrix operand order repaired and checked by reading, not compiled, not run on hardware
//
#include <hip/hip_runtime.h>

#define HW_ 65536
#define WID_ 256
#define NT 128
#define NPIX 131072

typedef __attribute__((ext_vector_type(16))) _Float16 v16h;
typedef __attribute__((ext_vector_type(8)))  float    v8f;

// --- CDNA5 async global->LDS copy helpers (ASYNCcnt path) -------------------
__device__ __forceinline__ void async_copy_b32(unsigned lds_addr, const void* gptr)
{
    asm volatile("global_load_async_to_lds_b32 %0, %1, off"
                 :: "v"(lds_addr), "v"((unsigned long long)(uintptr_t)gptr)
                 : "memory");
}
__device__ __forceinline__ void wait_async_all()
{
    asm volatile("s_wait_asynccnt 0" ::: "memory");
}
__device__ __forceinline__ unsigned lds_off(const void* p)
{
    // generic->LDS aperture: low 32 bits are the wave-relative LDS address
    return (unsigned)(uintptr_t)p;
}

// ---------------------------------------------------------------------------
// Fused (LN | LN*v | gelu-gate) + 1x1-conv GEMM using v_wmma_f32_16x16x32_f16.
// GEMM: out[COUT, pix] = W[COUT, CIN] x act[CIN, pix] per 128-pixel tile.
// MODE 0: act = LN(in) over CIN channels                      -> f16 out (LDS-staged, coalesced b128 stores)
// MODE 1: act = LN(in)*v                                      -> f32 out + resid
// MODE 2: act = gelu(in[k]) * in[k+CIN]   (in has 2*CIN ch)   -> f32 out + resid
// ---------------------------------------------------------------------------
template<int COUT, int CIN, int KPAD, int MODE, typename TIN>
__global__ __launch_bounds__(256)
void ffc_ln_gemm(const TIN* __restrict__ in, int inTotC,
                 const float* __restrict__ wgt,
                 const float* __restrict__ lnw, const float* __restrict__ lnb,
                 const _Float16* __restrict__ vsrc, int vTotC, int vOff,
                 const float* __restrict__ resid,
                 void* __restrict__ outp)
{
    __shared__ _Float16 sW[COUT * KPAD];                 // weights, K zero-padded
    __shared__ _Float16 sAct[KPAD * NT];                 // activations, K zero-padded
    __shared__ _Float16 sOut[(MODE == 0) ? COUT * NT : 1];
    const int tid  = threadIdx.x;
    const int tile = blockIdx.x;

    __builtin_prefetch(wgt, 0, 0);                       // global_prefetch_b8

    // stage weights (f32 -> f16), zero-pad K to KPAD
    for (int i = tid; i < COUT * KPAD; i += 256) {
        int m = i / KPAD, k = i % KPAD;
        sW[i] = (k < CIN) ? (_Float16)wgt[m * CIN + k] : (_Float16)0.f;
    }

    // stage activations: one thread per pixel of the tile
    if (tid < NT) {
        int g = tile * NT + tid;
        int b = g / HW_, p = g % HW_;
        const size_t ibase = (size_t)b * inTotC * HW_ + (size_t)p;
        if (MODE == 2) {
            for (int k = 0; k < KPAD; ++k) {
                float a = 0.f;
                if (k < CIN) {
                    float g1 = (float)in[ibase + (size_t)k * HW_];
                    float g2 = (float)in[ibase + (size_t)(k + CIN) * HW_];
                    a = 0.5f * g1 * (1.f + erff(g1 * 0.70710678118f)) * g2;
                }
                sAct[k * NT + tid] = (_Float16)a;
            }
        } else {
            float s = 0.f, ss = 0.f;
            for (int k = 0; k < CIN; ++k) {
                float v = (float)in[ibase + (size_t)k * HW_];
                s += v; ss += v * v;
            }
            float mu   = s / (float)CIN;
            float var  = ss / (float)CIN - mu * mu;
            float rstd = rsqrtf(var + 1e-5f);
            for (int k = 0; k < KPAD; ++k) {
                float a = 0.f;
                if (k < CIN) {
                    float v = (float)in[ibase + (size_t)k * HW_];
                    a = (v - mu) * rstd * lnw[k] + lnb[k];
                    if (MODE == 1)
                        a *= (float)vsrc[(size_t)b * vTotC * HW_ +
                                         (size_t)(vOff + k) * HW_ + p];
                }
                sAct[k * NT + tid] = (_Float16)a;
            }
        }
    }
    __syncthreads();

    const int wave = tid >> 5, lane = tid & 31;   // wave32
    const int MT = COUT / 16;
    for (int job = wave; job < MT * (NT / 16); job += 8) {
        int mt = job / (NT / 16), nt = job % (NT / 16);
        v8f acc = {};
        for (int k0 = 0; k0 < KPAD; k0 += 32) {
            v16h afr, bfr;
            // A fragment: 16x32 f16 layout (ISA 7.12.2)
            int m  = mt * 16 + (lane & 15);
            int kb = (lane < 16) ? 0 : 8;
            #pragma unroll
            for (int e = 0; e < 16; ++e) {
                int vg = e >> 1, pos = e & 1;
                int kk = k0 + ((vg < 4) ? (kb + vg * 2 + pos)
                                        : (16 + kb + (vg - 4) * 2 + pos));
                afr[e] = sW[m * KPAD + kk];
            }
            // B fragment: 32x16 f16, same K order as A: lanes 0-15 -> K 0..7,16..23, lanes 16-31 -> K 8..15,24..31
            int n   = nt * 16 + (lane & 15);
            int kbb = (lane < 16) ? 0 : 8;
            #pragma unroll
            for (int e = 0; e < 16; ++e)
                bfr[e] = sAct[(k0 + ((e < 8) ? (kbb + e) : (16 + kbb + (e - 8)))) * NT + n];
            acc = __builtin_amdgcn_wmma_f32_16x16x32_f16(
                false, afr, false, bfr, (short)0, acc, false, false);
        }
        // C/D layout: VGPR r -> M = r (lanes 0-15) / r+8 (lanes 16-31), N = lane&15
        int n = nt * 16 + (lane & 15);
        int mbase = mt * 16 + ((lane < 16) ? 0 : 8);
        if (MODE == 0) {
            #pragma unroll
            for (int r = 0; r < 8; ++r)
                sOut[(mbase + r) * NT + n] = (_Float16)acc[r];
        } else {
            int g = tile * NT + n;
            int b = g / HW_, p = g % HW_;
            #pragma unroll
            for (int r = 0; r < 8; ++r) {
                int m = mbase + r;
                size_t oidx = (size_t)b * COUT * HW_ + (size_t)m * HW_ + p;
                ((float*)outp)[oidx] = acc[r] + resid[oidx];
            }
        }
    }

    if (MODE == 0) {
        // cooperative coalesced copy-out: each row (channel) is 256B contiguous
        __syncthreads();
        int g0 = tile * NT;
        int b  = g0 / HW_, p0 = g0 % HW_;
        _Float16* outF16 = (_Float16*)outp;
        const uint4* src = (const uint4*)sOut;            // 16B = 8 f16
        for (int idx = tid; idx < COUT * (NT / 8); idx += 256) {
            int m = idx >> 4, cc = idx & 15;              // NT/8 = 16 chunks per row
            uint4 v = src[idx];
            *(uint4*)(outF16 + (size_t)b * COUT * HW_ + (size_t)m * HW_ +
                      (size_t)p0 + cc * 8) = v;
        }
    }
}

// ---------------------------------------------------------------------------
// Depthwise 3x3, SAME zero padding. One thread per (b, c, y, x).
// ---------------------------------------------------------------------------
__global__ __launch_bounds__(256)
void ffc_dwconv3(const _Float16* __restrict__ in, const float* __restrict__ w,
                 _Float16* __restrict__ out, int C)
{
    int i = blockIdx.x * 256 + threadIdx.x;      // grid sized exactly
    int p = i % HW_;
    int c = (i / HW_) % C;
    int y = p >> 8, x = p & 255;
    size_t base = (size_t)i - p;
    float acc = 0.f;
    #pragma unroll
    for (int dy = -1; dy <= 1; ++dy) {
        int yy = y + dy;
        if (yy < 0 || yy >= 256) continue;
        #pragma unroll
        for (int dx = -1; dx <= 1; ++dx) {
            int xx = x + dx;
            if (xx < 0 || xx >= 256) continue;
            acc += (float)in[base + (size_t)yy * WID_ + xx] *
                   w[c * 9 + (dy + 1) * 3 + (dx + 1)];
        }
    }
    out[i] = (_Float16)acc;
}

// ---------------------------------------------------------------------------
// FSAS: per-patch circular conv  out = irfft2(rfft2(q) * rfft2(k)) = q (*) k.
// 4 (channel,patch) jobs per 256-thread block; 64 threads per job.
// Patch staging uses global_load_async_to_lds_b32 (ASYNCcnt).
// ---------------------------------------------------------------------------
__global__ __launch_bounds__(256)
void ffc_circ_corr(const _Float16* __restrict__ qkv, _Float16* __restrict__ out)
{
    __shared__ _Float16 qs[4][64];
    __shared__ _Float16 ks[4][64];
    int tid = threadIdx.x;
    int grp = tid >> 6, t = tid & 63;
    int job = blockIdx.x * 4 + grp;              // total divisible by 4
    int pi = job % 1024;
    int c  = (job / 1024) % 96;
    int b  = job / (1024 * 96);
    int py = pi >> 5, px = pi & 31;

    // async stage: threads t<32 copy q as f16-pairs, t>=32 copy k as f16-pairs
    {
        int e  = t & 31;                          // pair index 0..31
        int m  = e >> 2, n2 = (e & 3) * 2;        // row, first col of the pair
        size_t pp   = (size_t)(py * 8 + m) * WID_ + (size_t)(px * 8 + n2);
        size_t gidx = (size_t)b * 288 * HW_ + (size_t)c * HW_ + pp;
        if (t < 32)
            async_copy_b32(lds_off(&qs[grp][m * 8 + n2]), qkv + gidx);
        else
            async_copy_b32(lds_off(&ks[grp][m * 8 + n2]),
                           qkv + gidx + (size_t)96 * HW_);
    }
    wait_async_all();
    __syncthreads();

    int m = t >> 3, n = t & 7;
    float acc = 0.f;
    #pragma unroll
    for (int i = 0; i < 8; ++i) {
        int mi = (m - i) & 7;
        #pragma unroll
        for (int j = 0; j < 8; ++j)
            acc += (float)qs[grp][i * 8 + j] * (float)ks[grp][mi * 8 + ((n - j) & 7)];
    }
    size_t p = (size_t)(py * 8 + m) * WID_ + (size_t)(px * 8 + n);
    out[(size_t)b * 96 * HW_ + (size_t)c * HW_ + p] = (_Float16)acc;
}

// ---------------------------------------------------------------------------
// Build spatial filter s_c = irfft2(ffn_fft_c)  (real weight -> cosine sum).
// All twiddles are cos(2*pi*r/8), r in [0,8).
// ---------------------------------------------------------------------------
__global__ void ffc_make_filter(const float* __restrict__ fw, float* __restrict__ s)
{
    int c = blockIdx.x;                          // 288 blocks x 64 threads
    int m = threadIdx.x >> 3, n = threadIdx.x & 7;
    const float ct[8] = {1.f, 0.70710678f, 0.f, -0.70710678f,
                         -1.f, -0.70710678f, 0.f, 0.70710678f};
    float acc = 0.f;
    for (int u = 0; u < 8; ++u)
        for (int v = 0; v < 8; ++v) {
            float f = (v <= 4) ? fw[(c * 8 + u) * 5 + v]
                               : fw[(c * 8 + ((8 - u) & 7)) * 5 + (8 - v)];
            acc += f * ct[(u * m + v * n) & 7];
        }
    s[c * 64 + threadIdx.x] = acc * (1.f / 64.f);
}

// ---------------------------------------------------------------------------
// DFFN: per-patch circular conv with fixed per-channel spatial filter.
// x staged async as f16 pairs; filter staged async as f32 words.
// ---------------------------------------------------------------------------
__global__ __launch_bounds__(256)
void ffc_circ_filter(const _Float16* __restrict__ in, const float* __restrict__ filt,
                     _Float16* __restrict__ out)
{
    __shared__ _Float16 qs[4][64];
    __shared__ float    ks[4][64];
    int tid = threadIdx.x;
    int grp = tid >> 6, t = tid & 63;
    int job = blockIdx.x * 4 + grp;
    int pi = job % 1024;
    int c  = (job / 1024) % 288;
    int b  = job / (1024 * 288);
    int py = pi >> 5, px = pi & 31;

    // async stage: every thread copies one f32 filter word; t<32 also copy x pairs
    async_copy_b32(lds_off(&ks[grp][t]), filt + c * 64 + t);
    if (t < 32) {
        int m = t >> 2, n2 = (t & 3) * 2;
        size_t pp = (size_t)(py * 8 + m) * WID_ + (size_t)(px * 8 + n2);
        async_copy_b32(lds_off(&qs[grp][m * 8 + n2]),
                       in + (size_t)b * 288 * HW_ + (size_t)c * HW_ + pp);
    }
    wait_async_all();
    __syncthreads();

    int m = t >> 3, n = t & 7;
    float acc = 0.f;
    #pragma unroll
    for (int i = 0; i < 8; ++i) {
        int mi = (m - i) & 7;
        #pragma unroll
        for (int j = 0; j < 8; ++j)
            acc += (float)qs[grp][i * 8 + j] * ks[grp][mi * 8 + ((n - j) & 7)];
    }
    size_t p = (size_t)(py * 8 + m) * WID_ + (size_t)(px * 8 + n);
    out[(size_t)b * 288 * HW_ + (size_t)c * HW_ + p] = (_Float16)acc;
}

// ---------------------------------------------------------------------------
extern "C" void kernel_launch(void* const* d_in, const int* in_sizes, int n_in,
                              void* d_out, int out_size, void* d_ws, size_t ws_size,
                              hipStream_t stream)
{
    const float* x    = (const float*)d_in[0];
    const float* n1w  = (const float*)d_in[1];
    const float* n1b  = (const float*)d_in[2];
    const float* awh  = (const float*)d_in[3];   // [288,48]
    const float* adw  = (const float*)d_in[4];   // [288,9]
    const float* anw  = (const float*)d_in[5];
    const float* anb  = (const float*)d_in[6];
    const float* awo  = (const float*)d_in[7];   // [48,96]
    const float* n2w  = (const float*)d_in[8];
    const float* n2b  = (const float*)d_in[9];
    const float* fwin = (const float*)d_in[10];  // [288,48]
    const float* ffft = (const float*)d_in[11];  // [288,8,5]
    const float* fdw  = (const float*)d_in[12];  // [288,9]
    const float* fwout= (const float*)d_in[13];  // [48,144]

    char* ws = (char*)d_ws;
    const size_t BUF = (size_t)288 * NPIX * sizeof(_Float16);   // 75.5 MB
    _Float16* buf1 = (_Float16*)ws;
    _Float16* buf2 = (_Float16*)(ws + BUF);
    float*    y1   = (float*)(ws + 2 * BUF);                    // residual stream, f32
    float*    filt = (float*)(ws + 2 * BUF + (size_t)48 * NPIX * sizeof(float));
    float*    out  = (float*)d_out;

    dim3 blk(256);
    const int tiles = NPIX / NT;                 // 1024

    // ---- FSAS branch ----
    // LN1 + W_hidden GEMM (288x48) -> hidden (buf1, f16)
    ffc_ln_gemm<288, 48, 64, 0, float><<<tiles, blk, 0, stream>>>(
        x, 48, awh, n1w, n1b, nullptr, 0, 0, nullptr, buf1);
    // depthwise 3x3 -> qkv (buf2)
    ffc_dwconv3<<<(2 * 288 * HW_) / 256, blk, 0, stream>>>(buf1, adw, buf2, 288);
    // per-patch circular conv q(*)k -> corr (buf1, 96ch layout)
    ffc_circ_corr<<<(2 * 96 * 1024) / 4, blk, 0, stream>>>(buf2, buf1);
    // LN(corr)*v + W_out GEMM (48x96) + residual x -> y1 (f32)
    ffc_ln_gemm<48, 96, 96, 1, _Float16><<<tiles, blk, 0, stream>>>(
        buf1, 96, awo, anw, anb, buf2, 288, 192, x, y1);

    // ---- DFFN branch ----
    // LN2 + W_in GEMM (288x48) -> t (buf2, f16)
    ffc_ln_gemm<288, 48, 64, 0, float><<<tiles, blk, 0, stream>>>(
        y1, 48, fwin, n2w, n2b, nullptr, 0, 0, nullptr, buf2);
    // spatial filter s_c = irfft2(ffn_fft_c)
    ffc_make_filter<<<288, 64, 0, stream>>>(ffft, filt);
    // per-patch circular conv with s_c -> tf (buf1)
    ffc_circ_filter<<<(2 * 288 * 1024) / 4, blk, 0, stream>>>(buf2, filt, buf1);
    // depthwise 3x3 -> g (buf2)
    ffc_dwconv3<<<(2 * 288 * HW_) / 256, blk, 0, stream>>>(buf1, fdw, buf2, 288);
    // gelu(g1)*g2 + W_out GEMM (48x144) + residual y1 -> d_out (f32)
    ffc_ln_gemm<48, 144, 160, 2, _Float16><<<tiles, blk, 0, stream>>>(
        buf2, 288, fwout, nullptr, nullptr, nullptr, 0, 0, y1, out);
}
